// EdgePredictor_31138512896412
// MI455X (gfx1250) — matrix operand order repaired and checked by reading, not compiled, not run on hardware
//
#include <hip/hip_runtime.h>

typedef unsigned short ushort_t;
typedef __attribute__((ext_vector_type(16))) __bf16 bf16x16;
typedef __attribute__((ext_vector_type(8)))  float  floatx8;

#define NNODES 10000
#define FIN    128
#define NEDGE  160000
#define ETOT   170000      // NEDGE + NNODES self loops
#define ELAB   100000
#define FEA    16
#define H1     4
#define CH     128
#define HC1    512         // H1*CH
#define KF     160         // padded feature K (144 -> 160)
#define NHID   64

__device__ __forceinline__ ushort_t f2bf(float f) {
  unsigned u = __float_as_uint(f);
  unsigned r = u + 0x7FFFu + ((u >> 16) & 1u);   // round-to-nearest-even
  return (ushort_t)(r >> 16);
}

__device__ __forceinline__ unsigned ford(float f) {   // order-preserving float->uint
  unsigned u = __float_as_uint(f);
  return (u & 0x80000000u) ? ~u : (u | 0x80000000u);
}
__device__ __forceinline__ float funord(unsigned o) {
  unsigned u = (o & 0x80000000u) ? (o & 0x7FFFFFFFu) : ~o;
  return __uint_as_float(u);
}

// ---------------- utility kernels ----------------
__global__ void k_zero32(unsigned* p, int n) {
  int i = blockIdx.x * blockDim.x + threadIdx.x;
  if (i < n) p[i] = 0u;
}

__global__ void k_f32_to_bf16(const float* __restrict__ in, ushort_t* __restrict__ out, int n) {
  int i = blockIdx.x * blockDim.x + threadIdx.x;
  if (i < n) out[i] = f2bf(in[i]);
}

// in: fp32 [K][Nn] row-major; out: bf16 [Nn][Kpad] (transposed, zero-padded K)
__global__ void k_transpose_bf16(const float* __restrict__ in, ushort_t* __restrict__ out,
                                 int K, int Nn, int Kpad) {
  int i = blockIdx.x * blockDim.x + threadIdx.x;
  int total = Nn * Kpad;
  if (i >= total) return;
  int n = i / Kpad, k = i % Kpad;
  out[i] = (k < K) ? f2bf(in[(size_t)k * Nn + n]) : (ushort_t)0;
}

// ---------------- bf16 WMMA GEMM: C[M][Nn] = A[M][K] * Bt[Nn][K]^T ----------------
// A row-major bf16, Bt = B transposed (column-major B) bf16. Optional bias+relu.
// block = 256 threads (8 waves); one 16x16 tile per wave, K-loop step 32.
__global__ void k_gemm_bf16(const ushort_t* __restrict__ A, const ushort_t* __restrict__ Bt,
                            float* __restrict__ Cmat, int M, int Nn, int K,
                            const float* __restrict__ bias, int relu) {
  const int wave = threadIdx.x >> 5;
  const int lane = threadIdx.x & 31;
  const int half = lane >> 4;
  const int l16  = lane & 15;
  const int ntiles = Nn >> 4;
  const int tile = blockIdx.x * (blockDim.x >> 5) + wave;
  if (tile >= (M >> 4) * ntiles) return;            // wave-uniform (EXEC stays full)
  const int mt = tile / ntiles;
  const int nt = tile % ntiles;
  const ushort_t* aRow = A  + (size_t)(mt * 16 + l16) * K;
  const ushort_t* bRow = Bt + (size_t)(nt * 16 + l16) * K;
  floatx8 acc = {};
  for (int k0 = 0; k0 < K; k0 += 32) {
    union { uint4 q[2]; bf16x16 v; } fa, fb;
    // A 16x32 layout: elems 0..7 -> k0+8*half .. +7 ; elems 8..15 -> k0+16+8*half .. +7
    fa.q[0] = *reinterpret_cast<const uint4*>(aRow + k0 + 8 * half);
    fa.q[1] = *reinterpret_cast<const uint4*>(aRow + k0 + 16 + 8 * half);
    // B 32x16 layout: lane column n, elems 0..7 -> k0+8*half .. +7 ; elems 8..15 -> k0+16+8*half .. +7
    fb.q[0] = *reinterpret_cast<const uint4*>(bRow + k0 + 8 * half);
    fb.q[1] = *reinterpret_cast<const uint4*>(bRow + k0 + 16 + 8 * half);
    acc = __builtin_amdgcn_wmma_f32_16x16x32_bf16(false, fa.v, false, fb.v,
                                                  (short)0, acc, false, false);
  }
  const int col = nt * 16 + l16;
  const float bv = bias ? bias[col] : 0.f;
  for (int r = 0; r < 8; ++r) {
    const int row = mt * 16 + r + 8 * half;          // C/D layout: M = r + 8*half
    float v = acc[r] + bv;
    if (relu) v = fmaxf(v, 0.f);
    Cmat[(size_t)row * Nn + col] = v;
  }
}

// ---------------- attention scores: a_src[n][h] = <h[n,h,:], att_src[h,:]> ----------------
// grid = NNODES blocks, block = H*32 (one wave per head)
__global__ void k_att_scores(const float* __restrict__ h, const float* __restrict__ att_s,
                             const float* __restrict__ att_d,
                             float* __restrict__ as, float* __restrict__ ad, int H, int Cc) {
  int node = blockIdx.x;
  int head = threadIdx.x >> 5;
  int lane = threadIdx.x & 31;
  const float* row = h + (size_t)node * H * Cc + (size_t)head * Cc;
  float ss = 0.f, sd = 0.f;
  for (int c = lane; c < Cc; c += 32) {
    float v = row[c];
    ss += v * att_s[head * Cc + c];
    sd += v * att_d[head * Cc + c];
  }
  for (int o = 16; o; o >>= 1) { ss += __shfl_xor(ss, o, 32); sd += __shfl_xor(sd, o, 32); }
  if (lane == 0) { as[node * H + head] = ss; ad[node * H + head] = sd; }
}

// ---------------- pass 1: alpha = leaky_relu(a_src[s]+a_dst[d]); segment max ----------------
__global__ void k_alpha_max(const float* __restrict__ as, const float* __restrict__ ad,
                            const int* __restrict__ ei, float* __restrict__ alpha,
                            unsigned* __restrict__ amax, int E, int H) {
  int idx = blockIdx.x * blockDim.x + threadIdx.x;
  int tot = (E + NNODES) * H;
  if (idx >= tot) return;
  int e = idx / H, hh = idx % H;
  int s, d;
  if (e < E) { s = ei[e]; d = ei[E + e]; } else { s = e - E; d = s; }
  float a = as[s * H + hh] + ad[d * H + hh];
  a = a > 0.f ? a : 0.2f * a;
  alpha[idx] = a;
  atomicMax(&amax[d * H + hh], ford(a));
}

// ---------------- pass 2: e = exp(alpha - max); segment sum ----------------
__global__ void k_exp_sum(const int* __restrict__ ei, float* __restrict__ alpha,
                          const unsigned* __restrict__ amax, float* __restrict__ denom,
                          int E, int H) {
  int idx = blockIdx.x * blockDim.x + threadIdx.x;
  int tot = (E + NNODES) * H;
  if (idx >= tot) return;
  int e = idx / H, hh = idx % H;
  int d = (e < E) ? ei[E + e] : (e - E);
  float m  = funord(amax[d * H + hh]);
  float ea = expf(alpha[idx] - m);
  alpha[idx] = ea;
  atomicAdd(&denom[d * H + hh], ea);
}

// ---------------- pass 3: agg[d] += h[s] * (e/denom[d]) ----------------
// grid = ETOT blocks
__global__ void k_aggregate(const float* __restrict__ hmat, const float* __restrict__ alpha,
                            const float* __restrict__ denom, const int* __restrict__ ei,
                            float* __restrict__ agg, int E, int H, int Cc) {
  int e = blockIdx.x;
  int s, d;
  if (e < E) { s = ei[e]; d = ei[E + e]; } else { s = e - E; d = s; }
  int HC = H * Cc;
  for (int hc = threadIdx.x; hc < HC; hc += blockDim.x) {
    int hh = hc / Cc;
    float coeff = alpha[e * H + hh] / fmaxf(denom[d * H + hh], 1e-16f);
    atomicAdd(&agg[(size_t)d * HC + hc], hmat[(size_t)s * HC + hc] * coeff);
  }
}

// ---------------- epilogue: elu(agg + bias) -> fp32 and/or bf16 ----------------
__global__ void k_epilogue(const float* __restrict__ agg, const float* __restrict__ bias,
                           float* __restrict__ out_f32, ushort_t* __restrict__ out_bf,
                           int total, int HC) {
  int i = blockIdx.x * blockDim.x + threadIdx.x;
  if (i >= total) return;
  float v = agg[i] + bias[i % HC];
  v = v > 0.f ? v : (expf(v) - 1.f);
  if (out_f32) out_f32[i] = v;
  if (out_bf)  out_bf[i]  = f2bf(v);
}

// ---------------- build edge features: [h2[s]*h2[d] | attr | zeros] as bf16 [ELAB][KF] ----
// grid = ELAB blocks, block = 160
__global__ void k_feats(const float* __restrict__ h2, const int* __restrict__ eli,
                        const float* __restrict__ attr, ushort_t* __restrict__ feats) {
  int e = blockIdx.x, t = threadIdx.x;
  int s = eli[e], d = eli[ELAB + e];
  float v;
  if (t < CH)            v = h2[(size_t)s * CH + t] * h2[(size_t)d * CH + t];
  else if (t < CH + FEA) v = attr[(size_t)e * FEA + (t - CH)];
  else                   v = 0.f;
  feats[(size_t)e * KF + t] = f2bf(v);
}

// ---------------- final: out[e][0..1] = hidden[e] @ mlp_w2 + mlp_b2 ----------------
// one wave per edge
__global__ void k_final(const float* __restrict__ hidden, const float* __restrict__ w2,
                        const float* __restrict__ b2, float* __restrict__ out) {
  int wid  = (blockIdx.x * blockDim.x + threadIdx.x) >> 5;
  int lane = threadIdx.x & 31;
  if (wid >= ELAB) return;
  const float* hrow = hidden + (size_t)wid * NHID;
  float h0 = hrow[lane], h1v = hrow[lane + 32];
  for (int j = 0; j < 2; ++j) {
    float p = h0 * w2[lane * 2 + j] + h1v * w2[(lane + 32) * 2 + j];
    for (int o = 16; o; o >>= 1) p += __shfl_xor(p, o, 32);
    if (lane == 0) out[(size_t)wid * 2 + j] = p + b2[j];
  }
}

static inline unsigned gridFor(size_t n, unsigned b) { return (unsigned)((n + b - 1) / b); }

extern "C" void kernel_launch(void* const* d_in, const int* in_sizes, int n_in,
                              void* d_out, int out_size, void* d_ws, size_t ws_size,
                              hipStream_t stream) {
  const float* x      = (const float*)d_in[0];
  const int*   ei     = (const int*)d_in[1];
  /* d_in[2] edge_attr: unused by reference */
  const int*   eli    = (const int*)d_in[3];
  const float* elattr = (const float*)d_in[4];
  const float* W1     = (const float*)d_in[5];
  const float* as1    = (const float*)d_in[6];
  const float* ad1    = (const float*)d_in[7];
  const float* b1     = (const float*)d_in[8];
  const float* W2     = (const float*)d_in[9];
  const float* as2    = (const float*)d_in[10];
  const float* ad2    = (const float*)d_in[11];
  const float* b2     = (const float*)d_in[12];
  const float* mw1    = (const float*)d_in[13];
  const float* mb1    = (const float*)d_in[14];
  const float* mw2    = (const float*)d_in[15];
  const float* mb2    = (const float*)d_in[16];
  float* out = (float*)d_out;

  char* ws = (char*)d_ws;
  size_t off = 0;
  auto alloc = [&](size_t bytes) { size_t o = off; off += (bytes + 255) & ~(size_t)255; return o; };

  // persistent small region
  size_t o_W1t   = alloc((size_t)HC1 * FIN * 2);       // bf16 [512][128]
  size_t o_W2t   = alloc((size_t)CH * HC1 * 2);        // bf16 [128][512]
  size_t o_MW1t  = alloc((size_t)NHID * KF * 2);       // bf16 [64][160]
  size_t o_asrc  = alloc((size_t)NNODES * H1 * 4);
  size_t o_adst  = alloc((size_t)NNODES * H1 * 4);
  size_t o_amax  = alloc((size_t)NNODES * H1 * 4);
  size_t o_den   = alloc((size_t)NNODES * H1 * 4);
  size_t o_alpha = alloc((size_t)ETOT * H1 * 4);
  size_t o_h2a   = alloc((size_t)NNODES * CH * 4);     // h2_act fp32
  size_t o_big   = off;
  // phase-overlapped big region (sequential reuse; all offsets 16B aligned)
  size_t o_xbf   = o_big + 0;           // bf16 [10000][128]   (phase A)
  size_t o_h1    = o_big + 2560000;     // fp32 [10000][512]   (phase A)
  size_t o_agg1  = o_big + 23040000;    // fp32 [10000][512]   (phase A)
  size_t o_h1abf = o_big + 43520000;    // bf16 [10000][512]   (phase A->B)
  size_t o_h2    = o_big + 0;           // fp32 [10000][128]   (phase B)
  size_t o_agg2  = o_big + 5120000;     // fp32 [10000][128]   (phase B)
  size_t o_feats = o_big + 0;           // bf16 [100000][160]  (phase C)
  size_t o_hid   = o_big + 32000000;    // fp32 [100000][64]   (phase C)
  (void)ws_size; (void)in_sizes; (void)n_in; (void)out_size;

  ushort_t* W1t   = (ushort_t*)(ws + o_W1t);
  ushort_t* W2t   = (ushort_t*)(ws + o_W2t);
  ushort_t* MW1t  = (ushort_t*)(ws + o_MW1t);
  float*    a_src = (float*)(ws + o_asrc);
  float*    a_dst = (float*)(ws + o_adst);
  unsigned* amax  = (unsigned*)(ws + o_amax);
  float*    den   = (float*)(ws + o_den);
  float*    alpha = (float*)(ws + o_alpha);
  float*    h2a   = (float*)(ws + o_h2a);
  ushort_t* xbf   = (ushort_t*)(ws + o_xbf);
  float*    h1    = (float*)(ws + o_h1);
  float*    agg1  = (float*)(ws + o_agg1);
  ushort_t* h1abf = (ushort_t*)(ws + o_h1abf);
  float*    h2    = (float*)(ws + o_h2);
  float*    agg2  = (float*)(ws + o_agg2);
  ushort_t* feats = (ushort_t*)(ws + o_feats);
  float*    hid   = (float*)(ws + o_hid);

  // weight prep (bf16, transposed, K-padded)
  k_transpose_bf16<<<gridFor((size_t)HC1 * FIN, 256), 256, 0, stream>>>(W1, W1t, FIN, HC1, FIN);
  k_transpose_bf16<<<gridFor((size_t)CH * HC1, 256), 256, 0, stream>>>(W2, W2t, HC1, CH, HC1);
  k_transpose_bf16<<<gridFor((size_t)NHID * KF, 256), 256, 0, stream>>>(mw1, MW1t, 144, NHID, KF);
  k_f32_to_bf16<<<gridFor((size_t)NNODES * FIN, 256), 256, 0, stream>>>(x, xbf, NNODES * FIN);

  // ---- GAT layer 1 ----
  k_gemm_bf16<<<(NNODES / 16) * (HC1 / 16) / 8, 256, 0, stream>>>(xbf, W1t, h1, NNODES, HC1, FIN, nullptr, 0);
  k_att_scores<<<NNODES, H1 * 32, 0, stream>>>(h1, as1, ad1, a_src, a_dst, H1, CH);
  k_zero32<<<gridFor((size_t)NNODES * H1, 256), 256, 0, stream>>>(amax, NNODES * H1);
  k_zero32<<<gridFor((size_t)NNODES * H1, 256), 256, 0, stream>>>((unsigned*)den, NNODES * H1);
  k_zero32<<<gridFor((size_t)NNODES * HC1, 256), 256, 0, stream>>>((unsigned*)agg1, NNODES * HC1);
  k_alpha_max<<<gridFor((size_t)ETOT * H1, 256), 256, 0, stream>>>(a_src, a_dst, ei, alpha, amax, NEDGE, H1);
  k_exp_sum<<<gridFor((size_t)ETOT * H1, 256), 256, 0, stream>>>(ei, alpha, amax, den, NEDGE, H1);
  k_aggregate<<<ETOT, 256, 0, stream>>>(h1, alpha, den, ei, agg1, NEDGE, H1, CH);
  k_epilogue<<<gridFor((size_t)NNODES * HC1, 256), 256, 0, stream>>>(agg1, b1, nullptr, h1abf, NNODES * HC1, HC1);

  // ---- GAT layer 2 (H=1) ----
  k_gemm_bf16<<<(NNODES / 16) * (CH / 16) / 8, 256, 0, stream>>>(h1abf, W2t, h2, NNODES, CH, HC1, nullptr, 0);
  k_att_scores<<<NNODES, 32, 0, stream>>>(h2, as2, ad2, a_src, a_dst, 1, CH);
  k_zero32<<<gridFor((size_t)NNODES, 256), 256, 0, stream>>>(amax, NNODES);
  k_zero32<<<gridFor((size_t)NNODES, 256), 256, 0, stream>>>((unsigned*)den, NNODES);
  k_zero32<<<gridFor((size_t)NNODES * CH, 256), 256, 0, stream>>>((unsigned*)agg2, NNODES * CH);
  k_alpha_max<<<gridFor((size_t)ETOT, 256), 256, 0, stream>>>(a_src, a_dst, ei, alpha, amax, NEDGE, 1);
  k_exp_sum<<<gridFor((size_t)ETOT, 256), 256, 0, stream>>>(ei, alpha, amax, den, NEDGE, 1);
  k_aggregate<<<ETOT, 128, 0, stream>>>(h2, alpha, den, ei, agg2, NEDGE, 1, CH);
  k_epilogue<<<gridFor((size_t)NNODES * CH, 256), 256, 0, stream>>>(agg2, b2, h2a, nullptr, NNODES * CH, CH);

  // ---- edge MLP ----
  k_feats<<<ELAB, KF, 0, stream>>>(h2a, eli, elattr, feats);
  k_gemm_bf16<<<(ELAB / 16) * (NHID / 16) / 8, 256, 0, stream>>>(feats, MW1t, hid, ELAB, NHID, KF, mb1, 1);
  k_final<<<ELAB / 8, 256, 0, stream>>>(hid, mw2, mb2, out);
}
